// InpaintCAModel_31791347925098
// MI455X (gfx1250) — matrix operand order repaired and checked by reading, not compiled, not run on hardware
//
#include <hip/hip_runtime.h>
#include <hip/hip_bf16.h>
#include <math.h>

// ---------------- problem constants (B=4, H=W=96, C=128, RATE=2, KSIZE=3) ----
#define BATCH   4
#define HH      96          // full-res H, W
#define C_CH    128
#define HS      48          // downsampled grid (96 / RATE)
#define LPATCH  2304        // HS*HS  (M, N and K of the score matrix)
#define K1      1152        // 3*3*128 : K of GEMM1
#define N2      2048        // 4*4*128 : N of GEMM2
#define SCALE   10.0f

typedef __attribute__((ext_vector_type(16))) __bf16 v16bf;
typedef __attribute__((ext_vector_type(8)))  float  v8f;

union FragU { uint4 u[2]; v16bf v; };

// ---------------- WMMA fragment loaders (wave32 layouts, CDNA5 ISA) --
// A (16x32 bf16, MxK): lane m=lane&15, grp=lane>>4.
//   halves 0..7  <- K = k0 + grp*8 .. +7
//   halves 8..15 <- K = k0 + 16 + grp*8 .. +7
__device__ __forceinline__ v16bf load_a_frag(const __bf16* A, int ldk, int row0,
                                             int k0, int lane) {
  int grp = (lane >> 4) & 1;
  int m   = lane & 15;
  const __bf16* p = A + (size_t)(row0 + m) * ldk + k0 + grp * 8;
  FragU f;
  f.u[0] = *(const uint4*)(p);
  f.u[1] = *(const uint4*)(p + 16);
  return f.v;
}

// B (32x16 bf16, KxN) with operand stored as N rows x K cols (i.e. B^T rows):
//   lane n=lane&15, grp=lane>>4; halves 0..7 <- K = k0 + grp*8 .. +7, halves 8..15 <- K = k0 + 16 + grp*8 .. +7
__device__ __forceinline__ v16bf load_b_frag(const __bf16* B, int ldk, int col0,
                                             int k0, int lane) {
  int grp = (lane >> 4) & 1;
  int n   = lane & 15;
  const __bf16* p = B + (size_t)(col0 + n) * ldk + k0 + grp * 8;
  FragU f;
  f.u[0] = *(const uint4*)(p);
  f.u[1] = *(const uint4*)(p + 16);
  return f.v;
}

// C/D 16x16 f32: VGPR r holds row grp*8+r, col = lane&15
__device__ __forceinline__ void store_tile(float* C, int ldn, int row0, int col0,
                                           int lane, v8f acc) {
  int grp = (lane >> 4) & 1;
  int n   = lane & 15;
  float* p = C + (size_t)(row0 + grp * 8) * ldn + col0 + n;
#pragma unroll
  for (int r = 0; r < 8; ++r) p[(size_t)r * ldn] = acc[r];
}

// ---------------- WMMA GEMM: C[M,N] = A[M,K] * B[N,K]^T, per-batch strides ---
// block = 128 threads = 4 waves; block tile 64x128.
// wave tile 32x64: 2 A-tiles x 4 B-tiles -> 8 WMMAs per 12 b128 loads,
// B fragments reused across both A rows in-register (raises FLOP/byte ~1.33x).
__global__ __launch_bounds__(128) void gemm_bf16_wmma(
    const __bf16* __restrict__ A, const __bf16* __restrict__ Bm,
    float* __restrict__ C, int Kdim, int ldc,
    long aBatch, long bBatch, long cBatch) {
  const int lane = threadIdx.x & 31;
  const int wave = threadIdx.x >> 5;
  const __bf16* Ab = A  + (size_t)blockIdx.z * aBatch;
  const __bf16* Bb = Bm + (size_t)blockIdx.z * bBatch;
  float*        Cb = C  + (size_t)blockIdx.z * cBatch;
  const int m0 = blockIdx.y * 64  + (wave >> 1) * 32;
  const int n0 = blockIdx.x * 128 + (wave & 1) * 64;

  v8f acc[2][4];
#pragma unroll
  for (int r = 0; r < 2; ++r)
#pragma unroll
    for (int cI = 0; cI < 4; ++cI) acc[r][cI] = (v8f){};

  for (int k0 = 0; k0 < Kdim; k0 += 32) {
    v16bf fa0 = load_a_frag(Ab, Kdim, m0,      k0, lane);
    v16bf fa1 = load_a_frag(Ab, Kdim, m0 + 16, k0, lane);
    v16bf fb0 = load_b_frag(Bb, Kdim, n0,      k0, lane);
    v16bf fb1 = load_b_frag(Bb, Kdim, n0 + 16, k0, lane);
    v16bf fb2 = load_b_frag(Bb, Kdim, n0 + 32, k0, lane);
    v16bf fb3 = load_b_frag(Bb, Kdim, n0 + 48, k0, lane);
    acc[0][0] = __builtin_amdgcn_wmma_f32_16x16x32_bf16(false, fa0, false, fb0,
                                                (short)0, acc[0][0], false, false);
    acc[0][1] = __builtin_amdgcn_wmma_f32_16x16x32_bf16(false, fa0, false, fb1,
                                                (short)0, acc[0][1], false, false);
    acc[0][2] = __builtin_amdgcn_wmma_f32_16x16x32_bf16(false, fa0, false, fb2,
                                                (short)0, acc[0][2], false, false);
    acc[0][3] = __builtin_amdgcn_wmma_f32_16x16x32_bf16(false, fa0, false, fb3,
                                                (short)0, acc[0][3], false, false);
    acc[1][0] = __builtin_amdgcn_wmma_f32_16x16x32_bf16(false, fa1, false, fb0,
                                                (short)0, acc[1][0], false, false);
    acc[1][1] = __builtin_amdgcn_wmma_f32_16x16x32_bf16(false, fa1, false, fb1,
                                                (short)0, acc[1][1], false, false);
    acc[1][2] = __builtin_amdgcn_wmma_f32_16x16x32_bf16(false, fa1, false, fb2,
                                                (short)0, acc[1][2], false, false);
    acc[1][3] = __builtin_amdgcn_wmma_f32_16x16x32_bf16(false, fa1, false, fb3,
                                                (short)0, acc[1][3], false, false);
  }
#pragma unroll
  for (int r = 0; r < 2; ++r)
#pragma unroll
    for (int cI = 0; cI < 4; ++cI)
      store_tile(Cb, ldc, m0 + r * 16, n0 + cI * 16, lane, acc[r][cI]);
}

// ---------------- prep: f-patch matrix A1[b][p][ (x*3+y)*128 + c ] -----------
__global__ __launch_bounds__(128) void prep_fpatches(const float* __restrict__ f,
                                                     __bf16* __restrict__ A1) {
  const int l = blockIdx.x, b = blockIdx.y, c = threadIdx.x;
  const int li = l / HS, lj = l % HS;
  __bf16* dst = A1 + ((size_t)b * LPATCH + l) * K1;
#pragma unroll
  for (int x = 0; x < 3; ++x)
#pragma unroll
    for (int y = 0; y < 3; ++y) {
      int i2 = li - 1 + x, j2 = lj - 1 + y;
      float v = 0.f;
      if (i2 >= 0 && i2 < HS && j2 >= 0 && j2 < HS)
        v = f[(((size_t)b * HH + 2 * i2) * HH + 2 * j2) * C_CH + c];
      dst[(x * 3 + y) * C_CH + c] = (__bf16)v;
    }
}

// ---------------- prep: L2-normalized b-patch matrix B1 ----------------------
__global__ __launch_bounds__(128) void prep_wpatches(const float* __restrict__ bs,
                                                     __bf16* __restrict__ B1) {
  const int l = blockIdx.x, b = blockIdx.y, c = threadIdx.x;
  const int li = l / HS, lj = l % HS;
  float vals[9];
  float ss = 0.f;
#pragma unroll
  for (int x = 0; x < 3; ++x)
#pragma unroll
    for (int y = 0; y < 3; ++y) {
      int i2 = li - 1 + x, j2 = lj - 1 + y;
      float v = 0.f;
      if (i2 >= 0 && i2 < HS && j2 >= 0 && j2 < HS)
        v = bs[(((size_t)b * HH + 2 * i2) * HH + 2 * j2) * C_CH + c];
      vals[x * 3 + y] = v;
      ss += v * v;
    }
  __shared__ float red[128];
  red[c] = ss;
  __syncthreads();
  for (int s = 64; s > 0; s >>= 1) {
    if (c < s) red[c] += red[c + s];
    __syncthreads();
  }
  const float inv = 1.0f / fmaxf(sqrtf(red[0]), 1e-4f);
  __bf16* dst = B1 + ((size_t)b * LPATCH + l) * K1;
#pragma unroll
  for (int k = 0; k < 9; ++k) dst[k * C_CH + c] = (__bf16)(vals[k] * inv);
}

// ---------------- prep: raw 4x4 patches, pre-transposed: RT[b][n][l] ---------
__global__ __launch_bounds__(128) void prep_rawT(const float* __restrict__ bs,
                                                 __bf16* __restrict__ RT) {
  const int l = blockIdx.x, b = blockIdx.y, c = threadIdx.x;
  const int li = l / HS, lj = l % HS;
  __bf16* base = RT + (size_t)b * N2 * LPATCH;
#pragma unroll
  for (int x = 0; x < 4; ++x)
#pragma unroll
    for (int y = 0; y < 4; ++y) {
      int i2 = 2 * li + x - 1, j2 = 2 * lj + y - 1;
      float v = 0.f;
      if (i2 >= 0 && i2 < HH && j2 >= 0 && j2 < HH)
        v = bs[(((size_t)b * HH + i2) * HH + j2) * C_CH + c];
      base[((size_t)((x * 4 + y) * C_CH + c)) * LPATCH + l] = (__bf16)v;
    }
}

// ---------------- prep: mm[l] = (patch mask sum == 0) ------------------------
__global__ void prep_mask(const float* __restrict__ mask, float* __restrict__ mm) {
  const int l = blockIdx.x * 256 + threadIdx.x;
  if (l >= LPATCH) return;
  const int li = l / HS, lj = l % HS;
  float s = 0.f;
#pragma unroll
  for (int x = 0; x < 3; ++x)
#pragma unroll
    for (int y = 0; y < 3; ++y) {
      int i2 = li - 1 + x, j2 = lj - 1 + y;
      if (i2 >= 0 && i2 < HS && j2 >= 0 && j2 < HS)
        s += mask[(2 * i2) * HH + 2 * j2];
    }
  mm[l] = (s == 0.0f) ? 1.0f : 0.0f;
}

// ---------------- fuse pass 1: diagonal 3-tap over flat (p, l), i-major ------
__global__ void fuse_diag_a(const float* __restrict__ in, float* __restrict__ out) {
  long t = (long)blockIdx.x * 256 + threadIdx.x;
  const int l = (int)(t % LPATCH);
  long r = t / LPATCH;
  const int p = (int)(r % LPATCH);
  const int b = (int)(r / LPATCH);
  const float* base = in + (size_t)b * LPATCH * LPATCH;
  float acc = 0.f;
#pragma unroll
  for (int d = -1; d <= 1; ++d) {
    int pp = p + d, ll = l + d;
    if (pp >= 0 && pp < LPATCH && ll >= 0 && ll < LPATCH)
      acc += base[(size_t)pp * LPATCH + ll];
  }
  out[t] = acc;
}

// ---------------- fuse pass 2: same, in j-major flattening -------------------
__global__ void fuse_diag_b(const float* __restrict__ in, float* __restrict__ out) {
  long t = (long)blockIdx.x * 256 + threadIdx.x;
  const int l = (int)(t % LPATCH);
  long r = t / LPATCH;
  const int p = (int)(r % LPATCH);
  const int b = (int)(r / LPATCH);
  const int i = p / HS, j = p % HS, li = l / HS, lj = l % HS;
  const int pj = j * HS + i, lt = lj * HS + li;  // j-major flat indices
  const float* base = in + (size_t)b * LPATCH * LPATCH;
  float acc = 0.f;
#pragma unroll
  for (int d = -1; d <= 1; ++d) {
    int q = pj + d, s2 = lt + d;
    if (q >= 0 && q < LPATCH && s2 >= 0 && s2 < LPATCH) {
      int i2 = q % HS, j2 = q / HS, li2 = s2 % HS, lj2 = s2 / HS;
      acc += base[(size_t)(i2 * HS + j2) * LPATCH + (li2 * HS + lj2)];
    }
  }
  out[t] = acc;
}

// ---------------- masked softmax over l, output bf16 for GEMM2 ---------------
__global__ __launch_bounds__(256) void softmax_rows(const float* __restrict__ in,
                                                    const float* __restrict__ mm,
                                                    __bf16* __restrict__ out) {
  const int p = blockIdx.x, b = blockIdx.y, tid = threadIdx.x;
  const float* row = in + ((size_t)b * LPATCH + p) * LPATCH;
  __bf16* orow = out + ((size_t)b * LPATCH + p) * LPATCH;
  __shared__ float red[256];
  float vloc[9], mloc[9];
  float mx = -1e30f;
#pragma unroll
  for (int t = 0; t < 9; ++t) {
    const int l = tid + t * 256;
    const float mk = mm[l];
    const float v = row[l] * mk * SCALE;
    vloc[t] = v;
    mloc[t] = mk;
    mx = fmaxf(mx, v);
  }
  red[tid] = mx;
  __syncthreads();
  for (int s = 128; s > 0; s >>= 1) {
    if (tid < s) red[tid] = fmaxf(red[tid], red[tid + s]);
    __syncthreads();
  }
  mx = red[0];
  __syncthreads();
  float sum = 0.f;
#pragma unroll
  for (int t = 0; t < 9; ++t) {
    vloc[t] = __expf(vloc[t] - mx);
    sum += vloc[t];
  }
  red[tid] = sum;
  __syncthreads();
  for (int s = 128; s > 0; s >>= 1) {
    if (tid < s) red[tid] += red[tid + s];
    __syncthreads();
  }
  const float inv = 1.0f / red[0];
#pragma unroll
  for (int t = 0; t < 9; ++t)
    orow[tid + t * 256] = (__bf16)(vloc[t] * inv * mloc[t]);
}

// ---------------- epilogue: overlap-add as race-free gather, /4 --------------
// buf[2i+x, 2j+y] += contrib[i,j,x,y]; y_out[h,w] = buf[h+1, w+1] / 4
//  => x in {(h+1)&1, +2}, i = (h+1-x)/2 (valid if 0<=i<48); same for w/j.
__global__ void gather_out(const float* __restrict__ contrib,
                           float* __restrict__ out) {
  long t = (long)blockIdx.x * 256 + threadIdx.x;
  const int c = (int)(t % C_CH);
  long r = t / C_CH;
  const int w = (int)(r % HH); r /= HH;
  const int h = (int)(r % HH);
  const int b = (int)(r / HH);
  const int xa = (h + 1) & 1, ia = (h + 1 - xa) >> 1, xb = xa + 2, ib = ia - 1;
  const int ya = (w + 1) & 1, ja = (w + 1 - ya) >> 1, yb = ya + 2, jb = ja - 1;
  const int xs[2] = {xa, xb}, is[2] = {ia, ib};
  const int ys[2] = {ya, yb}, js[2] = {ja, jb};
  const float* cb = contrib + (size_t)b * LPATCH * N2;
  float acc = 0.f;
#pragma unroll
  for (int u = 0; u < 2; ++u)
#pragma unroll
    for (int v = 0; v < 2; ++v) {
      const int i = is[u], j = js[v];
      if (i >= 0 && i < HS && j >= 0 && j < HS) {
        const int p = i * HS + j;
        const int n = (xs[u] * 4 + ys[v]) * C_CH + c;
        acc += cb[(size_t)p * N2 + n];
      }
    }
  out[t] = acc * 0.25f;
}

// -----------------------------------------------------------------------------
extern "C" void kernel_launch(void* const* d_in, const int* in_sizes, int n_in,
                              void* d_out, int out_size, void* d_ws, size_t ws_size,
                              hipStream_t stream) {
  const float* f    = (const float*)d_in[0];
  const float* bs   = (const float*)d_in[1];
  const float* mask = (const float*)d_in[2];
  float* out = (float*)d_out;
  char* ws = (char*)d_ws;

  // workspace layout (all region sizes are 256B-aligned by construction)
  const size_t szA  = (size_t)BATCH * LPATCH * K1 * 2;       // 21.2 MB  bf16
  const size_t szB  = szA;                                   // 21.2 MB  bf16
  const size_t szRT = (size_t)BATCH * N2 * LPATCH * 2;       // 37.7 MB  bf16
  const size_t szY  = (size_t)BATCH * LPATCH * LPATCH * 4;   // 84.9 MB  f32
  const size_t szYS = (size_t)BATCH * LPATCH * LPATCH * 2;   // 42.5 MB  bf16
  size_t off = 0;
  __bf16* A1 = (__bf16*)(ws + off); off += szA;
  __bf16* B1 = (__bf16*)(ws + off); off += szB;
  __bf16* RT = (__bf16*)(ws + off); off += szRT;
  float*  Y0 = (float*)(ws + off);  off += szY;   // scores ping
  float*  Y1 = (float*)(ws + off);  off += szY;   // scores pong; reused for contrib
  __bf16* YS = (__bf16*)(ws + off); off += szYS;  // softmaxed scores (bf16)
  float*  MM = (float*)(ws + off);  off += 256 * ((LPATCH * 4 + 255) / 256);
  float*  CONTRIB = Y1;  // 4*2304*2048*4 = 75.5 MB  <=  szY, Y1 is dead by then
  (void)ws_size; (void)in_sizes; (void)n_in; (void)out_size;

  // 1. prep operand matrices
  prep_mask<<<dim3((LPATCH + 255) / 256), 256, 0, stream>>>(mask, MM);
  prep_fpatches<<<dim3(LPATCH, BATCH), 128, 0, stream>>>(f, A1);
  prep_wpatches<<<dim3(LPATCH, BATCH), 128, 0, stream>>>(bs, B1);
  prep_rawT<<<dim3(LPATCH, BATCH), 128, 0, stream>>>(bs, RT);

  // 2. GEMM1: Y0[b] = A1[b](2304x1152) x B1[b]^T  -> 2304x2304 scores
  gemm_bf16_wmma<<<dim3(LPATCH / 128, LPATCH / 64, BATCH), 128, 0, stream>>>(
      A1, B1, Y0, K1, LPATCH,
      (long)LPATCH * K1, (long)LPATCH * K1, (long)LPATCH * LPATCH);

  // 3. fuse blurs (ping-pong Y0 -> Y1 -> Y0)
  const long nScore = (long)BATCH * LPATCH * LPATCH;
  fuse_diag_a<<<dim3((unsigned)(nScore / 256)), 256, 0, stream>>>(Y0, Y1);
  fuse_diag_b<<<dim3((unsigned)(nScore / 256)), 256, 0, stream>>>(Y1, Y0);

  // 4. masked softmax -> bf16 attention weights
  softmax_rows<<<dim3(LPATCH, BATCH), 256, 0, stream>>>(Y0, MM, YS);

  // 5. GEMM2: CONTRIB[b] = YS[b](2304x2304) x RT[b]^T -> 2304x2048
  gemm_bf16_wmma<<<dim3(N2 / 128, LPATCH / 64, BATCH), 128, 0, stream>>>(
      YS, RT, CONTRIB, LPATCH, N2,
      (long)LPATCH * LPATCH, (long)N2 * LPATCH, (long)LPATCH * N2);

  // 6. overlap-add epilogue (gather form), write output
  const long nOut = (long)BATCH * HH * HH * C_CH;
  gather_out<<<dim3((unsigned)(nOut / 256)), 256, 0, stream>>>(CONTRIB, out);
}
